// VertexingTask_86199993631146
// MI455X (gfx1250) — matrix operand order repaired and checked by reading, not compiled, not run on hardware
//
#include <hip/hip_runtime.h>
#include <hip/hip_bf16.h>

// ---------------------------------------------------------------------------
// VertexingTask: per-edge MLP over all ordered track pairs.
//   B=64 jets, N=64 tracks, D=128, ctx=128. Edges per jet e = 63*64 = 4032.
//   h1 = ReLU([x_i|x_j|ctx] @ W1 + b1)   (384 -> 256)
//   h2 = ReLU(h1 @ W2 + b2)              (256 -> 128)
//   out = h2 @ W3 + b3                   (128 -> 1)
// Restructured: h1(b,i,j) = ReLU(P[b,i] + Q[b,j] + R[b]) with
//   P = x @ W1[0:128], Q = x @ W1[128:256], R = ctx @ W1[256:384] + b1.
// Heavy part (h2 GEMM, ~17 GFLOP) uses v_wmma_f32_16x16x32_f16; LDS staging
// uses GLOBAL_LOAD_ASYNC_TO_LDS_B128 (ASYNCcnt) with double buffering.
// ---------------------------------------------------------------------------

typedef __attribute__((ext_vector_type(16))) _Float16 v16h;
typedef __attribute__((ext_vector_type(8)))  _Float16 h8;
typedef __attribute__((ext_vector_type(8)))  float    v8f;

#define LDS_PAD 8            // halfs of row padding (keeps 16B align, breaks bank conflicts)
#define NTRK    64
#define DTRK    128
#define H1DIM   256
#define H2DIM   128
#define EPJ     (NTRK * (NTRK - 1))   // 4032 edges per jet
#define NBN     (64 * NTRK)           // 4096 (b,n) rows

#if __has_builtin(__builtin_amdgcn_global_load_async_to_lds_b128) && \
    __has_builtin(__builtin_amdgcn_s_wait_asynccnt)
#define HAVE_ASYNC_LDS 1
#endif

#ifdef HAVE_ASYNC_LDS
// Builtin signature (from hipcc diagnostic): param0 = AS1 pointer to
// 'int __attribute__((vector_size(16)))', so match that pointee type exactly.
typedef int v4i __attribute__((vector_size(16)));
typedef __attribute__((address_space(1))) v4i as1_v4i;
typedef __attribute__((address_space(3))) v4i as3_v4i;
#endif

// 16B global -> LDS copy; async (ASYNCcnt) when available, else through VGPRs.
__device__ __forceinline__ void copy_b128(_Float16* dst_lds, const _Float16* src) {
#ifdef HAVE_ASYNC_LDS
  __builtin_amdgcn_global_load_async_to_lds_b128(
      (as1_v4i*)(uintptr_t)src,         // flat->AS1: full 64-bit address
      (as3_v4i*)(uintptr_t)dst_lds,     // flat LDS addr: low 32 bits = LDS offset
      0, 0);
#else
  *(h8*)dst_lds = *(const h8*)src;
#endif
}

__device__ __forceinline__ void wait_async_all() {
#ifdef HAVE_ASYNC_LDS
  __builtin_amdgcn_s_wait_asynccnt(0);
#endif
}

// ---- WMMA fragment loaders (CDNA5 16-bit A 16x32 / B 32x16 layouts) --------
// A (16x32, MxK), lane L: group = L>>4, m = L&15.
//   elements 0..7  = K  8*group + 0..7     (contiguous 16B)
//   elements 8..15 = K 16 + 8*group + 0..7 (contiguous 16B)
__device__ __forceinline__ v16h frag_a(const _Float16* base, int ldh, int ln) {
  const int grp = ln >> 4;
  const int m   = ln & 15;
  const _Float16* p = base + m * ldh + grp * 8;
  h8 lo = *(const h8*)(p);
  h8 hi = *(const h8*)(p + 16);
  return __builtin_shufflevector(lo, hi, 0,1,2,3,4,5,6,7,8,9,10,11,12,13,14,15);
}

// B (32x16, KxN), lane L: group = L>>4, n = L&15.
//   elements 0..7  = K  8*group + 0..7, elements 8..15 = K 16 + 8*group + 0..7
__device__ __forceinline__ v16h frag_b(const _Float16* base, int ldh, int ln) {
  const int grp = ln >> 4;
  const int n   = ln & 15;
  const _Float16* p = base + (grp * 8) * ldh + n;
  v16h v;
#pragma unroll
  for (int s = 0; s < 8; ++s) {
    v[s]     = p[s * ldh];
    v[8 + s] = p[(16 + s) * ldh];
  }
  return v;
}

// ---- conversion kernels ----------------------------------------------------
__global__ __launch_bounds__(256) void cvt_x_kernel(const float* __restrict__ x,
                                                    _Float16* __restrict__ xh, int n) {
  int idx = blockIdx.x * 256 + threadIdx.x;
  if (idx < n) xh[idx] = (_Float16)x[idx];
}

// W1cat[k][c] (128x512 f16): c<256 -> W1[k][c] (W1a), else W1[k+128][c-256] (W1b)
__global__ __launch_bounds__(256) void cvt_w1cat_kernel(const float* __restrict__ W1,
                                                        _Float16* __restrict__ W1cat) {
  int idx = blockIdx.x * 256 + threadIdx.x;       // 0 .. 128*512-1
  int k = idx >> 9, c = idx & 511;
  float v = (c < 256) ? W1[k * 256 + c] : W1[(k + 128) * 256 + (c - 256)];
  W1cat[idx] = (_Float16)v;
}

__global__ __launch_bounds__(256) void cvt_w2_kernel(const float* __restrict__ W2,
                                                     _Float16* __restrict__ W2h) {
  int idx = blockIdx.x * 256 + threadIdx.x;       // 0 .. 256*128-1
  W2h[idx] = (_Float16)W2[idx];
}

// R[b][n] = b1[n] + sum_k ctx[b][k] * W1[256+k][n]   (fp32, tiny)
__global__ __launch_bounds__(256) void ctx_proj_kernel(const float* __restrict__ ctx,
                                                       const float* __restrict__ W1,
                                                       const float* __restrict__ b1,
                                                       float* __restrict__ R) {
  int b = blockIdx.x, n = threadIdx.x;            // 64 blocks x 256 threads
  float acc = b1[n];
  const float* c = ctx + b * 128;
  for (int k = 0; k < 128; ++k)
    acc = fmaf(c[k], W1[(256 + k) * 256 + n], acc);
  R[b * 256 + n] = acc;
}

// ---- PQ GEMM: PQ[4096][512] = xh[4096][128] @ W1cat[128][512] (fp32 out) ---
__global__ __launch_bounds__(256) void pq_gemm_kernel(const _Float16* __restrict__ xh,
                                                      const _Float16* __restrict__ W1cat,
                                                      float* __restrict__ PQ) {
  __shared__ _Float16 sA[64][DTRK + LDS_PAD];     // 64x128 x tile
  __shared__ _Float16 sB[DTRK][128 + LDS_PAD];    // 128x128 W1cat tile

  const int mblk = blockIdx.x >> 2;               // 64 m-blocks of 64 rows
  const int nblk = blockIdx.x & 3;                // 4 n-blocks of 128 cols
  const int t = threadIdx.x;

  // stage A: 64 rows x 128 halfs = 1024 x b128 units
#pragma unroll
  for (int q = 0; q < 4; ++q) {
    int u = t * 4 + q;
    int row = u >> 4, col = (u & 15) * 8;
    copy_b128(&sA[row][col], xh + (mblk * 64 + row) * DTRK + col);
  }
  // stage B: 128 rows x 128 halfs = 2048 x b128 units
#pragma unroll
  for (int q = 0; q < 8; ++q) {
    int u = t * 8 + q;
    int row = u >> 4, col = (u & 15) * 8;
    copy_b128(&sB[row][col], W1cat + row * 512 + nblk * 128 + col);
  }
  wait_async_all();
  __syncthreads();

  const int w = t >> 5, ln = t & 31;
  const int mt = w & 3;                           // m-tile (16 rows)
  const int nhalf = w >> 2;                       // which 64-col half
  v8f acc[4];
#pragma unroll
  for (int nt = 0; nt < 4; ++nt) acc[nt] = (v8f){0,0,0,0,0,0,0,0};

#pragma unroll
  for (int kt = 0; kt < 4; ++kt) {                // K = 128 = 4 x 32
    v16h a = frag_a(&sA[mt * 16][kt * 32], DTRK + LDS_PAD, ln);
#pragma unroll
    for (int nt = 0; nt < 4; ++nt) {
      v16h bb = frag_b(&sB[kt * 32][nhalf * 64 + nt * 16], 128 + LDS_PAD, ln);
      acc[nt] = __builtin_amdgcn_wmma_f32_16x16x32_f16(
          false, a, false, bb, (short)0, acc[nt], false, false);
    }
  }

  // C layout: lane L: n = L&15 (+tile), rows = (L>>4)*8 + r (+tile)
  const int mrow = mblk * 64 + mt * 16 + (ln >> 4) * 8;
  const int ncol = nblk * 128 + nhalf * 64 + (ln & 15);
#pragma unroll
  for (int nt = 0; nt < 4; ++nt)
#pragma unroll
    for (int r = 0; r < 8; ++r)
      PQ[(mrow + r) * 512 + ncol + nt * 16] = acc[nt][r];
}

// ---- per-(b,i) edge MLP: h1 -> (WMMA) h2 -> dot(w3) ------------------------
// W2 staged in 8 chunks of 32 K-rows, double buffered via async LDS loads.
__global__ __launch_bounds__(256) void edge_mlp_kernel(const float* __restrict__ PQ,
                                                       const float* __restrict__ R,
                                                       const _Float16* __restrict__ W2h,
                                                       const float* __restrict__ b2,
                                                       const float* __restrict__ w3,
                                                       const float* __restrict__ b3,
                                                       float* __restrict__ out) {
  __shared__ _Float16 sH[NTRK][H1DIM + LDS_PAD];        // h1 rows (f16), ~33 KB
  __shared__ _Float16 sW2[2][32][H2DIM + LDS_PAD];      // double-buffered chunk, ~17 KB
  __shared__ float sOut[NTRK];

  const int blk = blockIdx.x;                     // 0..4095
  const int b = blk >> 6, i = blk & 63;
  const int t = threadIdx.x;

  // prefetch W2 chunk 0 while h1 is being built (32 rows x 128 halfs = 512 b128)
#pragma unroll
  for (int q = 0; q < 2; ++q) {
    int u = t * 2 + q;
    int row = u >> 4, col = (u & 15) * 8;
    copy_b128(&sW2[0][row][col], W2h + row * H2DIM + col);
  }

  if (t < NTRK) sOut[t] = 0.0f;

  { // build h1[j][k] = ReLU(P_i[k] + Q_j[k] + R_b[k]) -> f16 LDS
    const int j = t >> 2;
    const int k0 = (t & 3) * 64;
    const float* Pi = PQ + (b * 64 + i) * 512;          // cols [0,256)
    const float* Qj = PQ + (b * 64 + j) * 512 + 256;    // cols [256,512)
    const float* Rb = R + b * 256;
    for (int k = k0; k < k0 + 64; k += 4) {
      float4 p = *(const float4*)(Pi + k);
      float4 q = *(const float4*)(Qj + k);
      float4 r = *(const float4*)(Rb + k);
      sH[j][k + 0] = (_Float16)fmaxf(p.x + q.x + r.x, 0.0f);
      sH[j][k + 1] = (_Float16)fmaxf(p.y + q.y + r.y, 0.0f);
      sH[j][k + 2] = (_Float16)fmaxf(p.z + q.z + r.z, 0.0f);
      sH[j][k + 3] = (_Float16)fmaxf(p.w + q.w + r.w, 0.0f);
    }
  }

  const int w = t >> 5, ln = t & 31;
  const int mt = w & 3;                           // j-tile (16 rows)
  const int nhalf = w >> 2;                       // which 64 of the 128 h2 cols
  v8f acc[4];
#pragma unroll
  for (int nt = 0; nt < 4; ++nt) acc[nt] = (v8f){0,0,0,0,0,0,0,0};

#pragma unroll
  for (int kc = 0; kc < 8; ++kc) {                // K = 256 in 8 chunks of 32
    if (kc < 7) {                                 // prefetch next chunk
#pragma unroll
      for (int q = 0; q < 2; ++q) {
        int u = t * 2 + q;
        int row = u >> 4, col = (u & 15) * 8;
        copy_b128(&sW2[(kc + 1) & 1][row][col],
                  W2h + ((kc + 1) * 32 + row) * H2DIM + col);
      }
    }
#ifdef HAVE_ASYNC_LDS
    // in-order per-wave completion: <=2 outstanding => current chunk resident
    if (kc < 7) __builtin_amdgcn_s_wait_asynccnt(2);
    else        __builtin_amdgcn_s_wait_asynccnt(0);
#endif
    __syncthreads();                              // chunk kc visible to all waves

    v16h a = frag_a(&sH[mt * 16][kc * 32], H1DIM + LDS_PAD, ln);
#pragma unroll
    for (int nt = 0; nt < 4; ++nt) {
      v16h bb = frag_b(&sW2[kc & 1][0][nhalf * 64 + nt * 16], H2DIM + LDS_PAD, ln);
      acc[nt] = __builtin_amdgcn_wmma_f32_16x16x32_f16(
          false, a, false, bb, (short)0, acc[nt], false, false);
    }
    __syncthreads();                              // before buffer is overwritten
  }

  // epilogue: relu(h2 + b2) . w3, reduced per row j
  float w3v[4], b2v[4];
#pragma unroll
  for (int nt = 0; nt < 4; ++nt) {
    int n = nhalf * 64 + nt * 16 + (ln & 15);
    w3v[nt] = w3[n];
    b2v[nt] = b2[n];
  }
#pragma unroll
  for (int r = 0; r < 8; ++r) {
    float s = 0.0f;
#pragma unroll
    for (int nt = 0; nt < 4; ++nt)
      s = fmaf(fmaxf(acc[nt][r] + b2v[nt], 0.0f), w3v[nt], s);
    // butterfly reduce across the 16-lane group (wave32)
    s += __shfl_xor(s, 1, 32);
    s += __shfl_xor(s, 2, 32);
    s += __shfl_xor(s, 4, 32);
    s += __shfl_xor(s, 8, 32);
    if ((ln & 15) == 0)
      atomicAdd(&sOut[mt * 16 + (ln >> 4) * 8 + r], s);  // ds_add_f32
  }
  __syncthreads();

  if (t < NTRK) {
    int j = t;
    if (j != i) {
      int e = i * 63 + ((j < i) ? j : j - 1);     // row-major (i,j), i!=j
      out[b * EPJ + e] = sOut[j] + b3[0];
    }
  }
}

// ---------------------------------------------------------------------------
extern "C" void kernel_launch(void* const* d_in, const int* in_sizes, int n_in,
                              void* d_out, int out_size, void* d_ws, size_t ws_size,
                              hipStream_t stream) {
  const float* x   = (const float*)d_in[0];   // [64,64,128]
  const float* ctx = (const float*)d_in[1];   // [64,128]
  const float* W1  = (const float*)d_in[2];   // [384,256]
  const float* b1  = (const float*)d_in[3];   // [256]
  const float* W2  = (const float*)d_in[4];   // [256,128]
  const float* b2  = (const float*)d_in[5];   // [128]
  const float* W3  = (const float*)d_in[6];   // [128,1]
  const float* b3  = (const float*)d_in[7];   // [1]
  float* out = (float*)d_out;

  // workspace carve-up (all 16B aligned)
  char* ws = (char*)d_ws;
  size_t off = 0;
  _Float16* xh    = (_Float16*)(ws + off); off += (size_t)NBN * DTRK * 2;   // 1 MB
  _Float16* W1cat = (_Float16*)(ws + off); off += (size_t)128 * 512 * 2;    // 128 KB
  _Float16* W2h   = (_Float16*)(ws + off); off += (size_t)256 * 128 * 2;    // 64 KB
  float*    PQ    = (float*)(ws + off);    off += (size_t)NBN * 512 * 4;    // 8 MB
  float*    R     = (float*)(ws + off);    off += (size_t)64 * 256 * 4;     // 64 KB
  (void)off; (void)ws_size; (void)in_sizes; (void)n_in; (void)out_size;

  cvt_x_kernel<<<(NBN * DTRK) / 256, 256, 0, stream>>>(x, xh, NBN * DTRK);
  cvt_w1cat_kernel<<<(128 * 512) / 256, 256, 0, stream>>>(W1, W1cat);
  cvt_w2_kernel<<<(256 * 128) / 256, 256, 0, stream>>>(W2, W2h);
  ctx_proj_kernel<<<64, 256, 0, stream>>>(ctx, W1, b1, R);

  pq_gemm_kernel<<<256, 256, 0, stream>>>(xh, W1cat, PQ);
  edge_mlp_kernel<<<4096, 256, 0, stream>>>(PQ, R, W2h, b2, W3, b3, out);
}
